// get_model_70403103916615
// MI455X (gfx1250) — matrix operand order repaired and checked by reading, not compiled, not run on hardware
//
#include <hip/hip_runtime.h>

// ---------------------------------------------------------------------------
// WMMA helpers (CDNA5 / gfx1250, wave32, v_wmma_f32_16x16x32_f16)
// ---------------------------------------------------------------------------
typedef __attribute__((ext_vector_type(16))) _Float16 v16h;
typedef __attribute__((ext_vector_type(8)))  float    v8f;

union Frag { v16h h; unsigned u[8]; };

// A fragment: 16x32 f16.  lane = hf*16 + m.  VGPR j<4 : K = 8*hf + 2j
//                                            VGPR j>=4: K = 16 + 8*hf + 2(j-4)
__device__ __forceinline__ void load_fragA(Frag& f, const _Float16* rowBase, int hf) {
  const unsigned* p = (const unsigned*)rowBase;
  const int o = 4 * hf;
#pragma unroll
  for (int j = 0; j < 4; ++j) f.u[j] = p[o + j];
#pragma unroll
  for (int j = 0; j < 4; ++j) f.u[4 + j] = p[8 + o + j];
}

// B fragment: 32x16 f16. lane = hf*16 + n.  VGPR j<4 : K = 8*hf + 2j
//                                           VGPR j>=4: K = 16 + 8*hf + 2(j-4)
__device__ __forceinline__ void load_fragB(Frag& f, const _Float16* rowBase, int hf) {
  const unsigned* p = (const unsigned*)rowBase;
  const int o = 4 * hf;
#pragma unroll
  for (int j = 0; j < 4; ++j) f.u[j] = p[o + j];
#pragma unroll
  for (int j = 0; j < 4; ++j) f.u[4 + j] = p[8 + o + j];
}

__device__ __forceinline__ v8f wmma16(const Frag& a, const Frag& b, v8f c) {
  return __builtin_amdgcn_wmma_f32_16x16x32_f16(false, a.h, false, b.h,
                                                (short)0, c, false, false);
}

__device__ __forceinline__ int pad32d(int x) { return (x + 31) & ~31; }

// ---------------------------------------------------------------------------
// 1. transpose (B,3,N) -> (B,N,3)
// ---------------------------------------------------------------------------
__global__ void transpose_xyz_kernel(const float* __restrict__ xyz,
                                     float* __restrict__ out, int Bc, int n) {
  int t = blockIdx.x * 256 + threadIdx.x;
  int total = Bc * n * 3;
  if (t >= total) return;
  int b = t / (n * 3);
  int rem = t - b * n * 3;
  int i = rem / 3, k = rem - i * 3;
  out[t] = xyz[((size_t)b * 3 + k) * n + i];
}

// ---------------------------------------------------------------------------
// 2. farthest point sampling: one block (256 thr) per batch
// ---------------------------------------------------------------------------
__global__ void __launch_bounds__(256)
fps_kernel(const float* __restrict__ xyz, int n, int np, int* __restrict__ out) {
  __shared__ float sdist[2048];
  __shared__ float rv[256];
  __shared__ int   ri[256];
  int b = blockIdx.x, tid = threadIdx.x;
  const float* base = xyz + (size_t)b * n * 3;
  for (int j = tid; j < n; j += 256) sdist[j] = 1e10f;
  __syncthreads();
  int far = 0;
  for (int it = 0; it < np; ++it) {
    if (tid == 0) out[b * np + it] = far;
    float cx = base[far * 3 + 0], cy = base[far * 3 + 1], cz = base[far * 3 + 2];
    float bv = -1.f; int bi = 0;
    for (int j = tid; j < n; j += 256) {
      float dx = base[j * 3 + 0] - cx;
      float dy = base[j * 3 + 1] - cy;
      float dz = base[j * 3 + 2] - cz;
      float d = dx * dx + dy * dy + dz * dz;
      float nd = fminf(sdist[j], d);
      sdist[j] = nd;
      if (nd > bv) { bv = nd; bi = j; }
    }
    rv[tid] = bv; ri[tid] = bi;
    __syncthreads();
    for (int s = 128; s > 0; s >>= 1) {
      if (tid < s) {
        if (rv[tid + s] > rv[tid] ||
            (rv[tid + s] == rv[tid] && ri[tid + s] < ri[tid])) {
          rv[tid] = rv[tid + s]; ri[tid] = ri[tid + s];
        }
      }
      __syncthreads();
    }
    far = ri[0];
    __syncthreads();
  }
}

// ---------------------------------------------------------------------------
// 3. gather sampled coordinates
// ---------------------------------------------------------------------------
__global__ void gather_xyz_kernel(const float* __restrict__ xyz,
                                  const int* __restrict__ idx,
                                  float* __restrict__ out, int n, int S, int total) {
  int t = blockIdx.x * 256 + threadIdx.x;
  if (t >= total) return;
  int b = t / S;
  int gi = idx[t];
  const float* p = xyz + ((size_t)b * n + gi) * 3;
  out[t * 3 + 0] = p[0];
  out[t * 3 + 1] = p[1];
  out[t * 3 + 2] = p[2];
}

// ---------------------------------------------------------------------------
// 4. ball query: one wave per (b,s); ordered compaction via ballot+popcount
// ---------------------------------------------------------------------------
__global__ void __launch_bounds__(256)
ball_query_kernel(const float* __restrict__ xyz, int n,
                  const float* __restrict__ newxyz, int S, int total,
                  float r2, int K, int* __restrict__ gidx) {
  int wave = blockIdx.x * 8 + (threadIdx.x >> 5);
  if (wave >= total) return;
  int lane = threadIdx.x & 31;
  int b = wave / S;
  const float* q = newxyz + (size_t)wave * 3;
  float qx = q[0], qy = q[1], qz = q[2];
  const float* base = xyz + (size_t)b * n * 3;
  int* out = gidx + (size_t)wave * K;
  int cnt = 0, first = -1;
  for (int bj = 0; bj < n && cnt < K; bj += 32) {
    int j = bj + lane;
    float dx = base[j * 3 + 0] - qx;
    float dy = base[j * 3 + 1] - qy;
    float dz = base[j * 3 + 2] - qz;
    bool in = (dx * dx + dy * dy + dz * dz) <= r2;
    unsigned m = __builtin_amdgcn_ballot_w32(in);
    if (first < 0 && m) first = bj + __builtin_ctz(m);
    int pos = cnt + __builtin_popcount(m & ((1u << lane) - 1u));
    if (in && pos < K) out[pos] = j;
    cnt += __builtin_popcount(m);
  }
  if (cnt > K) cnt = K;
  if (first < 0) first = 0;
  for (int p = cnt + lane; p < K; p += 32) out[p] = first;
}

// ---------------------------------------------------------------------------
// 5. weight conversion f32 -> zero-padded f16 (cout32 x cin32), bias pad
// ---------------------------------------------------------------------------
__global__ void convert_w_kernel(const float* __restrict__ W,
                                 _Float16* __restrict__ W16,
                                 int cout, int cin, int cout32, int cin32) {
  int t = blockIdx.x * 256 + threadIdx.x;
  int total = cout32 * cin32;
  if (t >= total) return;
  int r = t / cin32, c = t - r * cin32;
  float v = (r < cout && c < cin) ? W[(size_t)r * cin + c] : 0.f;
  W16[t] = (_Float16)v;
}

__global__ void pad_bias_kernel(const float* __restrict__ bsrc,
                                float* __restrict__ bdst, int cout, int cout32) {
  int t = blockIdx.x * 256 + threadIdx.x;
  if (t >= cout32) return;
  bdst[t] = (t < cout) ? bsrc[t] : 0.f;
}

// ---------------------------------------------------------------------------
// 6. fused SA branch: gather -> 3-layer WMMA MLP in LDS -> max over K
// ---------------------------------------------------------------------------
__device__ __forceinline__ void lds_mlp_layer(const _Float16* src, int lds,
                                              _Float16* dst, int ldd,
                                              const _Float16* __restrict__ W,
                                              const float* __restrict__ bias,
                                              int cout32, int cinPad, int K) {
  int wave = threadIdx.x >> 5, lane = threadIdx.x & 31;
  int hf = lane >> 4, ln = lane & 15;
  int mt = K >> 4, nt = cout32 >> 4;
  for (int t = wave; t < mt * nt; t += 8) {
    int mi = t / nt, ni = t - mi * nt;
    v8f acc = {};
    const _Float16* arow = src + (size_t)(mi * 16 + ln) * lds;
    const _Float16* brow = W + (size_t)(ni * 16 + ln) * cinPad;
    for (int kk = 0; kk < cinPad; kk += 32) {
      Frag a, b;
      load_fragA(a, arow + kk, hf);
      load_fragB(b, brow + kk, hf);
      acc = wmma16(a, b, acc);
    }
    float bv = bias[ni * 16 + ln];
    int col = ni * 16 + ln;
#pragma unroll
    for (int r = 0; r < 8; ++r) {
      int row = mi * 16 + hf * 8 + r;
      float v = acc[r] + bv;
      v = v > 0.f ? v : 0.f;
      dst[(size_t)row * ldd + col] = (_Float16)v;
    }
  }
}

extern __shared__ char dynsmem[];

__global__ void __launch_bounds__(256)
sa_branch_kernel(const float* __restrict__ xyz, int n,
                 const float* __restrict__ pts, int cp,
                 const float* __restrict__ newxyz, int S,
                 const int* __restrict__ gidx, int K, int cin32,
                 const _Float16* W0, const float* b0, int co0,
                 const _Float16* W1, const float* b1, int co1,
                 const _Float16* W2, const float* b2, int co2,
                 int strideA, int strideB,
                 float* __restrict__ outPts, int ctot, int chOff) {
  _Float16* bufA = (_Float16*)dynsmem;
  _Float16* bufB = bufA + (size_t)K * strideA;
  int bs = blockIdx.x;
  int b = bs / S, s = bs - b * S;
  int tid = threadIdx.x;
  const int* g = gidx + (size_t)bs * K;
  const float* nx = newxyz + (size_t)bs * 3;
  int cinRaw = cp + 3;
  // gather grouped features into LDS (K x cin32, f16, zero pad)
  for (int t = tid; t < K * cin32; t += 256) {
    int k = t / cin32, c = t - k * cin32;
    float v = 0.f;
    int gi = g[k];
    if (c < cp)           v = pts[((size_t)b * n + gi) * cp + c];
    else if (c < cinRaw)  v = xyz[((size_t)b * n + gi) * 3 + (c - cp)] - nx[c - cp];
    bufA[(size_t)k * strideA + c] = (_Float16)v;
  }
  __syncthreads();
  int c0p = pad32d(co0), c1p = pad32d(co1), c2p = pad32d(co2);
  lds_mlp_layer(bufA, strideA, bufB, strideB, W0, b0, c0p, cin32, K);
  __syncthreads();
  lds_mlp_layer(bufB, strideB, bufA, strideA, W1, b1, c1p, c0p, K);
  __syncthreads();
  lds_mlp_layer(bufA, strideA, bufB, strideB, W2, b2, c2p, c1p, K);
  __syncthreads();
  // max over K samples
  for (int c = tid; c < co2; c += 256) {
    float m = -1e30f;
    for (int k = 0; k < K; ++k) {
      float v = (float)bufB[(size_t)k * strideB + c];
      m = v > m ? v : m;
    }
    outPts[((size_t)b * S + s) * ctot + chOff + c] = m;
  }
}

// ---------------------------------------------------------------------------
// 7. generic row GEMM: Y = act(X * W^T + b), one 16x16 tile per wave
// ---------------------------------------------------------------------------
__global__ void __launch_bounds__(256)
gemm_rows_kernel(const _Float16* __restrict__ X, int ldx,
                 const _Float16* __restrict__ W, const float* __restrict__ bias,
                 _Float16* __restrict__ Y, int ldy, float* __restrict__ Yf32,
                 int R, int cinPad, int cout32, int doRelu) {
  int wave = blockIdx.x * 8 + (threadIdx.x >> 5);
  int nt = cout32 >> 4;
  int tiles = (R >> 4) * nt;
  if (wave >= tiles) return;
  int mi = wave / nt, ni = wave - mi * nt;
  int lane = threadIdx.x & 31, hf = lane >> 4, ln = lane & 15;
  v8f acc = {};
  const _Float16* arow = X + (size_t)(mi * 16 + ln) * ldx;
  const _Float16* brow = W + (size_t)(ni * 16 + ln) * cinPad;
  for (int kk = 0; kk < cinPad; kk += 32) {
    Frag a, b;
    load_fragA(a, arow + kk, hf);
    load_fragB(b, brow + kk, hf);
    acc = wmma16(a, b, acc);
  }
  float bv = bias[ni * 16 + ln];
  int col = ni * 16 + ln;
#pragma unroll
  for (int r = 0; r < 8; ++r) {
    int row = mi * 16 + hf * 8 + r;
    float v = acc[r] + bv;
    if (doRelu) v = v > 0.f ? v : 0.f;
    Y[(size_t)row * ldy + col] = (_Float16)v;
    if (Yf32) Yf32[(size_t)row * cout32 + col] = v;
  }
}

// ---------------------------------------------------------------------------
// 8. input builders for the row GEMMs
// ---------------------------------------------------------------------------
__global__ void build_sa3_x_kernel(const float* __restrict__ l2xyz,
                                   const float* __restrict__ l2pts,
                                   _Float16* __restrict__ X) {
  int t = blockIdx.x * 256 + threadIdx.x;
  const int total = 2048 * 544;
  if (t >= total) return;
  int row = t / 544, c = t - row * 544;
  float v = 0.f;
  if (c < 3)        v = l2xyz[row * 3 + c];
  else if (c < 515) v = l2pts[(size_t)row * 512 + (c - 3)];
  X[t] = (_Float16)v;
}

__global__ void build_fp3_x_kernel(const float* __restrict__ l2pts,
                                   const float* __restrict__ l3pts,
                                   _Float16* __restrict__ X) {
  int t = blockIdx.x * 256 + threadIdx.x;
  const int total = 2048 * 1536;
  if (t >= total) return;
  int row = t / 1536, c = t - row * 1536;
  int b = row >> 7;
  float v = (c < 512) ? l2pts[(size_t)row * 512 + c]
                      : l3pts[(size_t)b * 1024 + (c - 512)];
  X[t] = (_Float16)v;
}

__global__ void build_fp2_x_kernel(const float* __restrict__ l1pts,
                                   const float* __restrict__ l2p,
                                   const int* __restrict__ idx3,
                                   const float* __restrict__ w3,
                                   _Float16* __restrict__ X) {
  int t = blockIdx.x * 256 + threadIdx.x;
  const int total = 8192 * 576;
  if (t >= total) return;
  int row = t / 576, c = t - row * 576;
  int b = row >> 9;
  float v;
  if (c < 320) v = l1pts[(size_t)row * 320 + c];
  else {
    int cc = c - 320;
    v = 0.f;
#pragma unroll
    for (int k = 0; k < 3; ++k) {
      int id = idx3[row * 3 + k];
      v += w3[row * 3 + k] * l2p[((size_t)b * 128 + id) * 256 + cc];
    }
  }
  X[t] = (_Float16)v;
}

__global__ void build_fp1_x_kernel(const float* __restrict__ cls,
                                   const float* __restrict__ l0xyz,
                                   const float* __restrict__ l1p,
                                   const int* __restrict__ idx3,
                                   const float* __restrict__ w3,
                                   _Float16* __restrict__ X) {
  int t = blockIdx.x * 256 + threadIdx.x;
  const int total = 32768 * 160;
  if (t >= total) return;
  int row = t / 160, c = t - row * 160;
  int b = row >> 11;
  float v = 0.f;
  if (c < 16)       v = cls[b * 16 + c];
  else if (c < 19)  v = l0xyz[(size_t)row * 3 + (c - 16)];
  else if (c < 22)  v = l0xyz[(size_t)row * 3 + (c - 19)];
  else if (c < 150) {
    int cc = c - 22;
#pragma unroll
    for (int k = 0; k < 3; ++k) {
      int id = idx3[row * 3 + k];
      v += w3[row * 3 + k] * l1p[((size_t)b * 512 + id) * 128 + cc];
    }
  }
  X[t] = (_Float16)v;
}

// ---------------------------------------------------------------------------
// 9. 3-NN interpolation weights
// ---------------------------------------------------------------------------
__global__ void three_nn_kernel(const float* __restrict__ qxyz, int nq,
                                const float* __restrict__ rxyz, int nr,
                                int total, int* __restrict__ idx3,
                                float* __restrict__ w3) {
  int t = blockIdx.x * 256 + threadIdx.x;
  if (t >= total) return;
  int b = t / nq;
  const float* q = qxyz + (size_t)t * 3;
  const float* rb = rxyz + (size_t)b * nr * 3;
  float d0 = 1e30f, d1 = 1e30f, d2 = 1e30f;
  int i0 = 0, i1 = 0, i2 = 0;
  float qx = q[0], qy = q[1], qz = q[2];
  for (int j = 0; j < nr; ++j) {
    float dx = rb[j * 3 + 0] - qx;
    float dy = rb[j * 3 + 1] - qy;
    float dz = rb[j * 3 + 2] - qz;
    float d = dx * dx + dy * dy + dz * dz;
    if (d < d0)      { d2 = d1; i2 = i1; d1 = d0; i1 = i0; d0 = d; i0 = j; }
    else if (d < d1) { d2 = d1; i2 = i1; d1 = d; i1 = j; }
    else if (d < d2) { d2 = d; i2 = j; }
  }
  float r0 = 1.f / (d0 + 1e-8f), r1 = 1.f / (d1 + 1e-8f), r2 = 1.f / (d2 + 1e-8f);
  float s = r0 + r1 + r2;
  idx3[t * 3 + 0] = i0; idx3[t * 3 + 1] = i1; idx3[t * 3 + 2] = i2;
  w3[t * 3 + 0] = r0 / s; w3[t * 3 + 1] = r1 / s; w3[t * 3 + 2] = r2 / s;
}

// ---------------------------------------------------------------------------
// 10. misc: max over rows, f16->f32, transposes, log_softmax
// ---------------------------------------------------------------------------
__global__ void max_rows_kernel(const _Float16* __restrict__ Y,
                                float* __restrict__ out, int groups, int K, int C) {
  int t = blockIdx.x * 256 + threadIdx.x;
  if (t >= groups * C) return;
  int b = t / C, c = t - b * C;
  float m = -1e30f;
  for (int k = 0; k < K; ++k) {
    float v = (float)Y[((size_t)b * K + k) * C + c];
    m = v > m ? v : m;
  }
  out[t] = m;
}

__global__ void h2f_kernel(const _Float16* __restrict__ src,
                           float* __restrict__ dst, int n) {
  int t = blockIdx.x * 256 + threadIdx.x;
  if (t < n) dst[t] = (float)src[t];
}

__global__ void transpose_f32_kernel(const float* __restrict__ src,
                                     float* __restrict__ dst, int Bc, int n, int C) {
  int t = blockIdx.x * 256 + threadIdx.x;
  int total = Bc * n * C;
  if (t >= total) return;
  int b = t / (n * C);
  int rem = t - b * n * C;
  int i = rem / C, c = rem - i * C;
  dst[((size_t)b * C + c) * n + i] = src[t];
}

__global__ void transpose_f16_kernel(const _Float16* __restrict__ src,
                                     float* __restrict__ dst, int Bc, int n, int C) {
  int t = blockIdx.x * 256 + threadIdx.x;
  int total = Bc * n * C;
  if (t >= total) return;
  int b = t / (n * C);
  int rem = t - b * n * C;
  int i = rem / C, c = rem - i * C;
  dst[((size_t)b * C + c) * n + i] = (float)src[t];
}

__global__ void log_softmax_kernel(const float* __restrict__ logits,
                                   float* __restrict__ out, int rows, int C, int ld) {
  int t = blockIdx.x * 256 + threadIdx.x;
  if (t >= rows) return;
  const float* p = logits + (size_t)t * ld;
  float m = -1e30f;
  for (int c = 0; c < C; ++c) m = fmaxf(m, p[c]);
  float s = 0.f;
  for (int c = 0; c < C; ++c) s += expf(p[c] - m);
  float ls = logf(s);
  float* o = out + (size_t)t * C;
  for (int c = 0; c < C; ++c) o[c] = p[c] - m - ls;
}

// ---------------------------------------------------------------------------
// host orchestration
// ---------------------------------------------------------------------------
static inline int pad32(int x) { return (x + 31) & ~31; }

extern "C" void kernel_launch(void* const* d_in, const int* in_sizes, int n_in,
                              void* d_out, int out_size, void* d_ws, size_t ws_size,
                              hipStream_t stream) {
  (void)in_sizes; (void)n_in; (void)out_size; (void)ws_size;
  const int Bb = 16, Nn = 2048, S1 = 512, S2 = 128;

  // ---- parameter flattening (top dict insertion order, params sorted keys) --
  struct LayerDesc { int wIdx, bIdx, cout, cin; };
  static const LayerDesc L[26] = {
    {2, 3, 128, 150}, {4, 5, 128, 128},                    // fp1
    {6, 7, 256, 576}, {8, 9, 128, 256},                    // fp2
    {10, 11, 256, 1536}, {12, 13, 256, 256},               // fp3
    {14, 15, 128, 128},                                    // head1
    {16, 17, 50, 128},                                     // head2
    {18, 19, 32, 6}, {20, 21, 32, 32}, {22, 23, 64, 32},   // sa1 b1 (8..10)
    {24, 25, 64, 6}, {26, 27, 64, 64}, {28, 29, 128, 64},  // sa1 b2 (11..13)
    {30, 31, 64, 6}, {32, 33, 96, 64}, {34, 35, 128, 96},  // sa1 b3 (14..16)
    {36, 37, 128, 323}, {38, 39, 128, 128}, {40, 41, 256, 128}, // sa2 b1 (17..19)
    {42, 43, 128, 323}, {44, 45, 196, 128}, {46, 47, 256, 196}, // sa2 b2 (20..22)
    {48, 49, 256, 515}, {50, 51, 512, 256}, {52, 53, 1024, 512} // sa3    (23..25)
  };

  char* wsBase = (char*)d_ws;
  size_t off = 0;
  auto alloc = [&](size_t bytes) -> void* {
    void* p = wsBase + off;
    off += (bytes + 255) & ~(size_t)255;
    return p;
  };

  float* l0xyz = (float*)alloc((size_t)Bb * Nn * 3 * 4);
  int*   fps1  = (int*)alloc((size_t)Bb * S1 * 4);
  float* nx1   = (float*)alloc((size_t)Bb * S1 * 3 * 4);
  int*   g1a   = (int*)alloc((size_t)Bb * S1 * 32 * 4);
  int*   g1b   = (int*)alloc((size_t)Bb * S1 * 64 * 4);
  int*   g1c   = (int*)alloc((size_t)Bb * S1 * 128 * 4);
  float* l1pts = (float*)alloc((size_t)Bb * S1 * 320 * 4);
  int*   fps2  = (int*)alloc((size_t)Bb * S2 * 4);
  float* nx2   = (float*)alloc((size_t)Bb * S2 * 3 * 4);
  int*   g2a   = (int*)alloc((size_t)Bb * S2 * 64 * 4);
  int*   g2b   = (int*)alloc((size_t)Bb * S2 * 128 * 4);
  float* l2pts = (float*)alloc((size_t)Bb * S2 * 512 * 4);
  float* l3pts = (float*)alloc((size_t)Bb * 1024 * 4);
  float* l2p   = (float*)alloc((size_t)Bb * S2 * 256 * 4);
  float* l1p   = (float*)alloc((size_t)Bb * S1 * 128 * 4);
  int*   idx3a = (int*)alloc((size_t)Bb * S1 * 3 * 4);
  float* w3a   = (float*)alloc((size_t)Bb * S1 * 3 * 4);
  int*   idx3b = (int*)alloc((size_t)Bb * Nn * 3 * 4);
  float* w3b   = (float*)alloc((size_t)Bb * Nn * 3 * 4);
  float* logits = (float*)alloc((size_t)Bb * Nn * 64 * 4);
  _Float16* GA = (_Float16*)alloc((size_t)6291456 * 2);
  _Float16* GB = (_Float16*)alloc((size_t)4194304 * 2);

  // ---- convert weights to padded f16 ----
  const _Float16* w16p[26];
  const float* bp[26];
  for (int i = 0; i < 26; ++i) {
    int c32 = pad32(L[i].cout), k32 = pad32(L[i].cin);
    _Float16* wp = (_Float16*)alloc((size_t)c32 * k32 * 2);
    float* bb = (float*)alloc((size_t)c32 * 4);
    int tot = c32 * k32;
    convert_w_kernel<<<(tot + 255) / 256, 256, 0, stream>>>(
        (const float*)d_in[L[i].wIdx], wp, L[i].cout, L[i].cin, c32, k32);
    pad_bias_kernel<<<(c32 + 255) / 256, 256, 0, stream>>>(
        (const float*)d_in[L[i].bIdx], bb, L[i].cout, c32);
    w16p[i] = wp; bp[i] = bb;
  }

  const float* xyzIn = (const float*)d_in[0];
  const float* clsIn = (const float*)d_in[1];
  float* outF = (float*)d_out;
  float* outX     = outF;                 // (B,N,50)
  float* outL1pT  = outF + 1638400;       // (B,128,512)
  float* outL2pT  = outF + 2686976;       // (B,256,128)
  float* outL3    = outF + 3211264;       // (B,1024,1)
  float* outFeatT = outF + 3227648;       // (B,128,2048)

  auto gemm = [&](const _Float16* X, int ldx, int li, _Float16* Y, float* Yf32,
                  int R, int relu) {
    int cinPad = pad32(L[li].cin), cout32 = pad32(L[li].cout);
    int tiles = (R / 16) * (cout32 / 16);
    gemm_rows_kernel<<<(tiles + 7) / 8, 256, 0, stream>>>(
        X, ldx, w16p[li], bp[li], Y, cout32, Yf32, R, cinPad, cout32, relu);
  };

  auto saBranch = [&](const float* xyz, int n, const float* pts, int cp,
                      const float* nxz, int S, const int* gi, int K,
                      int l0i, float* outPts, int ctot, int chOff) {
    int cin32 = pad32(cp + 3);
    int c0 = L[l0i].cout, c1 = L[l0i + 1].cout, c2 = L[l0i + 2].cout;
    int sA = cin32 > pad32(c1) ? cin32 : pad32(c1);
    int sB = pad32(c0) > pad32(c2) ? pad32(c0) : pad32(c2);
    size_t sh = (size_t)K * (sA + sB) * 2;
    sa_branch_kernel<<<Bb * S, 256, sh, stream>>>(
        xyz, n, pts, cp, nxz, S, gi, K, cin32,
        w16p[l0i], bp[l0i], c0,
        w16p[l0i + 1], bp[l0i + 1], c1,
        w16p[l0i + 2], bp[l0i + 2], c2,
        sA, sB, outPts, ctot, chOff);
  };

  // ---- l0 ----
  transpose_xyz_kernel<<<(Bb * Nn * 3 + 255) / 256, 256, 0, stream>>>(
      xyzIn, l0xyz, Bb, Nn);

  // ---- sa1 ----
  fps_kernel<<<Bb, 256, 0, stream>>>(l0xyz, Nn, S1, fps1);
  gather_xyz_kernel<<<(Bb * S1 + 255) / 256, 256, 0, stream>>>(
      l0xyz, fps1, nx1, Nn, S1, Bb * S1);
  ball_query_kernel<<<(Bb * S1 + 7) / 8, 256, 0, stream>>>(
      l0xyz, Nn, nx1, S1, Bb * S1, 0.01f, 32, g1a);
  ball_query_kernel<<<(Bb * S1 + 7) / 8, 256, 0, stream>>>(
      l0xyz, Nn, nx1, S1, Bb * S1, 0.04f, 64, g1b);
  ball_query_kernel<<<(Bb * S1 + 7) / 8, 256, 0, stream>>>(
      l0xyz, Nn, nx1, S1, Bb * S1, 0.16f, 128, g1c);
  saBranch(l0xyz, Nn, l0xyz, 3, nx1, S1, g1a, 32,  8,  l1pts, 320, 0);
  saBranch(l0xyz, Nn, l0xyz, 3, nx1, S1, g1b, 64,  11, l1pts, 320, 64);
  saBranch(l0xyz, Nn, l0xyz, 3, nx1, S1, g1c, 128, 14, l1pts, 320, 192);

  // ---- sa2 ----
  fps_kernel<<<Bb, 256, 0, stream>>>(nx1, S1, S2, fps2);
  gather_xyz_kernel<<<(Bb * S2 + 255) / 256, 256, 0, stream>>>(
      nx1, fps2, nx2, S1, S2, Bb * S2);
  ball_query_kernel<<<(Bb * S2 + 7) / 8, 256, 0, stream>>>(
      nx1, S1, nx2, S2, Bb * S2, 0.16f, 64, g2a);
  ball_query_kernel<<<(Bb * S2 + 7) / 8, 256, 0, stream>>>(
      nx1, S1, nx2, S2, Bb * S2, 0.64f, 128, g2b);
  saBranch(nx1, S1, l1pts, 320, nx2, S2, g2a, 64,  17, l2pts, 512, 0);
  saBranch(nx1, S1, l1pts, 320, nx2, S2, g2b, 128, 20, l2pts, 512, 256);

  // ---- sa3 (group-all, row GEMMs) ----
  build_sa3_x_kernel<<<(2048 * 544 + 255) / 256, 256, 0, stream>>>(nx2, l2pts, GA);
  gemm(GA, 544, 23, GB, nullptr, 2048, 1);   // 544 -> 256
  gemm(GB, 256, 24, GA, nullptr, 2048, 1);   // 256 -> 512
  gemm(GA, 512, 25, GB, nullptr, 2048, 1);   // 512 -> 1024
  max_rows_kernel<<<(Bb * 1024 + 255) / 256, 256, 0, stream>>>(GB, l3pts, Bb, 128, 1024);
  hipMemcpyAsync(outL3, l3pts, (size_t)Bb * 1024 * 4, hipMemcpyDeviceToDevice, stream);

  // ---- fp3 ----
  build_fp3_x_kernel<<<(2048 * 1536 + 255) / 256, 256, 0, stream>>>(l2pts, l3pts, GA);
  gemm(GA, 1536, 4, GB, nullptr, 2048, 1);
  gemm(GB, 256, 5, GA, nullptr, 2048, 1);
  h2f_kernel<<<(2048 * 256 + 255) / 256, 256, 0, stream>>>(GA, l2p, 2048 * 256);
  transpose_f32_kernel<<<(Bb * S2 * 256 + 255) / 256, 256, 0, stream>>>(
      l2p, outL2pT, Bb, S2, 256);

  // ---- fp2 ----
  three_nn_kernel<<<(Bb * S1 + 255) / 256, 256, 0, stream>>>(
      nx1, S1, nx2, S2, Bb * S1, idx3a, w3a);
  build_fp2_x_kernel<<<(8192 * 576 + 255) / 256, 256, 0, stream>>>(
      l1pts, l2p, idx3a, w3a, GA);
  gemm(GA, 576, 2, GB, nullptr, 8192, 1);
  gemm(GB, 256, 3, GA, nullptr, 8192, 1);
  h2f_kernel<<<(8192 * 128 + 255) / 256, 256, 0, stream>>>(GA, l1p, 8192 * 128);
  transpose_f32_kernel<<<(Bb * S1 * 128 + 255) / 256, 256, 0, stream>>>(
      l1p, outL1pT, Bb, S1, 128);

  // ---- fp1 + heads ----
  three_nn_kernel<<<(Bb * Nn + 255) / 256, 256, 0, stream>>>(
      l0xyz, Nn, nx1, S1, Bb * Nn, idx3b, w3b);
  build_fp1_x_kernel<<<(32768 * 160 + 255) / 256, 256, 0, stream>>>(
      clsIn, l0xyz, l1p, idx3b, w3b, GA);
  gemm(GA, 160, 0, GB, nullptr, 32768, 1);   // fp1 L0: 150 -> 128
  gemm(GB, 128, 1, GA, nullptr, 32768, 1);   // fp1 L1: 128 -> 128 (l0p)
  gemm(GA, 128, 6, GB, nullptr, 32768, 1);   // head1: 128 -> 128 (feat)
  transpose_f16_kernel<<<(Bb * Nn * 128 + 255) / 256, 256, 0, stream>>>(
      GB, outFeatT, Bb, Nn, 128);
  gemm(GB, 128, 7, GA, logits, 32768, 0);    // head2: 128 -> 50 (pad 64)
  log_softmax_kernel<<<(32768 + 255) / 256, 256, 0, stream>>>(
      logits, outX, 32768, 50, 64);
}
